// EdgeExpert_4612794876346
// MI455X (gfx1250) — matrix operand order repaired and checked by reading, not compiled, not run on hardware
//
#include <hip/hip_runtime.h>
#include <hip/hip_bf16.h>

typedef __attribute__((ext_vector_type(16))) __bf16 v16bf;
typedef __attribute__((ext_vector_type(8)))  float  v8f;

constexpr int Bn = 4, Cch = 512, Hh = 128, Ww = 128;
constexpr int Cq = 128;
constexpr int NKB = Cch / 32;            // 16 K-blocks of 32 channels

__device__ __forceinline__ float gelu_exact(float x) {
    return 0.5f * x * (1.0f + erff(x * 0.70710678118654752440f));
}

__device__ __forceinline__ unsigned lds_off(const void* p) {
    // generic shared pointer: low 32 bits are the LDS byte offset (flat rule)
    return (unsigned)(size_t)p;
}

// ---------------------------------------------------------------------------
// Pack branch weights [4][128][512] f32 -> bf16 A-fragment layout.
// A-frag order used here (contiguous k, same as B): lane<16: K in {0..15}, lane>=16: {16..31},
// row M = lane%16, VGPR v packs (k, k+1).
// ---------------------------------------------------------------------------
__global__ void pack_wbranch(const float* __restrict__ wb, __bf16* __restrict__ pw) {
    int fid = blockIdx.x * blockDim.x + threadIdx.x;   // 16384 lane-fragments
    int l  = fid & 31;
    int kb = (fid >> 5) & 15;
    int m  = (fid >> 9) & 7;
    int k  = fid >> 12;
    int o  = m * 16 + (l & 15);
    v16bf frag;
#pragma unroll
    for (int v = 0; v < 8; ++v) {
        int kl = 2 * v + ((l >= 16) ? 16 : 0);
        const float* src = wb + ((size_t)(k * Cq + o) * Cch + kb * 32 + kl);
        frag[2 * v]     = (__bf16)src[0];
        frag[2 * v + 1] = (__bf16)src[1];
    }
    *(v16bf*)(pw + (size_t)fid * 16) = frag;
}

// ---------------------------------------------------------------------------
// Pack w_fuse [512][512] with per-batch attention folded in -> bf16 A-frags.
// ---------------------------------------------------------------------------
__global__ void pack_wfuse(const float* __restrict__ wf, const float* __restrict__ att,
                           __bf16* __restrict__ pw) {
    int fid = blockIdx.x * blockDim.x + threadIdx.x;   // 65536 lane-fragments
    int l  = fid & 31;
    int kb = (fid >> 5) & 15;
    int m  = (fid >> 9) & 31;
    int b  = fid >> 14;
    int o  = m * 16 + (l & 15);
    v16bf frag;
#pragma unroll
    for (int v = 0; v < 8; ++v) {
        int kl = 2 * v + ((l >= 16) ? 16 : 0);
        int c  = kb * 32 + kl;
        frag[2 * v]     = (__bf16)(wf[(size_t)o * Cch + c]     * att[b * Cch + c]);
        frag[2 * v + 1] = (__bf16)(wf[(size_t)o * Cch + c + 1] * att[b * Cch + c + 1]);
    }
    *(v16bf*)(pw + (size_t)fid * 16) = frag;
}

// ---------------------------------------------------------------------------
// Pass 1: fused depthwise edge stencils + 4 branch GEMMs + BN + GELU.
// Grid 1024 (b,h,seg), block 512 = 16 waves; each wave owns 32 out-channels of
// one branch across a 64-pixel row tile.  x halo staged with
// GLOBAL_LOAD_ASYNC_TO_LDS (ASYNCcnt path); ef written as bf16 B-fragments
// via GLOBAL_STORE_ASYNC_FROM_LDS.
// ---------------------------------------------------------------------------
__global__ __launch_bounds__(512)
void edge_branch_gemm(const float* __restrict__ x,
                      const __bf16* __restrict__ pwb,
                      const float* __restrict__ b_branch,
                      const float* __restrict__ bn_gamma,
                      const float* __restrict__ bn_beta,
                      __bf16* __restrict__ ef,
                      float* __restrict__ partials) {
    __shared__ float  xs[3 * 32 * 68];          // halo tile, 32 channels
    __shared__ __bf16 featB[4 * 64 * 32];       // [branch][pixel][ch] bf16
    __shared__ float  scaleS[512], shiftS[512]; // folded BN scale / shift
    __shared__ __bf16 efT[64 * 528];            // [pixel][channel] (padded)

    const int tid  = threadIdx.x;
    const int wid  = blockIdx.x;
    const int b    = wid >> 8;
    const int h    = (wid >> 1) & 127;
    const int w0   = (wid & 1) * 64;
    const int wv   = tid >> 5;
    const int l    = tid & 31;
    const int lh   = l & 15;
    const int half = l >> 4;
    const int k    = wv >> 2;          // branch 0..3
    const int mt   = (wv & 3) * 2;     // first 16-row M-tile within branch

    {   // fold BN (eval mode, mean=0,var=1) into scale/shift per channel
        float s = bn_gamma[tid] * rsqrtf(1.0f + 1e-5f);
        scaleS[tid] = s;
        shiftS[tid] = b_branch[tid] * s + bn_beta[tid];
    }

    v8f acc[2][4] = {};
    const size_t xbase = (size_t)b * Cch * (Hh * Ww);

    for (int kb = 0; kb < NKB; ++kb) {
        __syncthreads();
        // stage x halo: 3 rows x 66 cols x 32 channels.
        // valid lanes: async copy memory -> LDS (no VGPR round trip);
        // zero-pad lanes: plain LDS store of 0.
        for (int i = tid; i < 3 * 32 * 66; i += 512) {
            int wi = i % 66;
            int rc = i / 66;
            int c  = rc & 31;
            int r  = rc >> 5;
            int gw = w0 - 1 + wi;
            int gh = h - 1 + r;
            if ((unsigned)gw < 128u && (unsigned)gh < 128u) {
                const float* gp = x + xbase + ((size_t)(kb * 32 + c) * Hh + gh) * Ww + gw;
                unsigned lo = lds_off(&xs[rc * 68 + wi]);
                asm volatile("global_load_async_to_lds_b32 %0, %1, off"
                             :: "v"(lo), "v"(gp) : "memory");
            } else {
                xs[rc * 68 + wi] = 0.0f;
            }
        }
        asm volatile("s_wait_asynccnt 0x0" ::: "memory");
        __syncthreads();
        // edge maps for 32ch x 64px (K_D2 == -K_D1 so diag = |d1|)
        for (int i = tid; i < 32 * 64; i += 512) {
            int c = i >> 6;
            int n = i & 63;
            const float* p0 = &xs[(0 * 32 + c) * 68 + n];
            const float* p1 = &xs[(1 * 32 + c) * 68 + n];
            const float* p2 = &xs[(2 * 32 + c) * 68 + n];
            float t00 = p0[0], t01 = p0[1], t02 = p0[2];
            float t10 = p1[0], t11 = p1[1], t12 = p1[2];
            float t20 = p2[0], t21 = p2[1], t22 = p2[2];
            float sx = (t02 - t00) + 2.0f * (t12 - t10) + (t22 - t20);
            float sy = (t20 - t00) + 2.0f * (t21 - t01) + (t22 - t02);
            float lp = t01 + t10 - 4.0f * t11 + t12 + t21;
            float d1 = t00 - t02 - t20 + t22;
            float s2 = sx * sx + sy * sy + 1e-8f;
            int idx = n * 32 + c;
            featB[0 * 2048 + idx] = (__bf16)sqrtf(s2);
            featB[1 * 2048 + idx] = (__bf16)fabsf(lp);
            featB[2 * 2048 + idx] = (__bf16)fabsf(d1);
            featB[3 * 2048 + idx] = (__bf16)sqrtf(s2 + lp * lp + 1e-8f);
        }
        __syncthreads();
        // 8 WMMAs per wave: 2 M-tiles x 4 N-tiles, K=32
        v16bf a0 = *(const v16bf*)(pwb + ((size_t)(((k * 8 + mt    ) * 16 + kb) * 32 + l)) * 16);
        v16bf a1 = *(const v16bf*)(pwb + ((size_t)(((k * 8 + mt + 1) * 16 + kb) * 32 + l)) * 16);
#pragma unroll
        for (int nt = 0; nt < 4; ++nt) {
            v16bf bf = *(const v16bf*)(&featB[k * 2048 + (nt * 16 + lh) * 32 + (half << 4)]);
            acc[0][nt] = __builtin_amdgcn_wmma_f32_16x16x32_bf16(false, a0, false, bf,
                             (short)0, acc[0][nt], false, false);
            acc[1][nt] = __builtin_amdgcn_wmma_f32_16x16x32_bf16(false, a1, false, bf,
                             (short)0, acc[1][nt], false, false);
        }
    }
    __syncthreads();

    // epilogue: BN + exact GELU, pooled row sums (deterministic shuffle tree),
    // transpose into LDS for fragment repacking
#pragma unroll
    for (int tm = 0; tm < 2; ++tm) {
#pragma unroll
        for (int r = 0; r < 8; ++r) {
            int grow = k * 128 + (mt + tm) * 16 + r + (half << 3);
            float s = scaleS[grow], t = shiftS[grow];
            float rowsum = 0.0f;
#pragma unroll
            for (int nt = 0; nt < 4; ++nt) {
                float g = gelu_exact(acc[tm][nt][r] * s + t);
                rowsum += g;
                efT[(nt * 16 + lh) * 528 + grow] = (__bf16)g;
            }
            rowsum += __shfl_xor(rowsum, 1);
            rowsum += __shfl_xor(rowsum, 2);
            rowsum += __shfl_xor(rowsum, 4);
            rowsum += __shfl_xor(rowsum, 8);
            if (lh == 0)
                partials[(size_t)wid * 512 + grow] = rowsum;
        }
    }
    __syncthreads();
    // repack bf16 ef into WMMA B-fragment order [tile][kb][lane][16] and
    // push straight from LDS to memory with async stores (IOFFSET applies to
    // both LDS and global address per ISA pseudocode).
    const int baseTile = ((b * Hh + h) * Ww + w0) >> 4;
    for (int f = tid; f < 16 * 4 * 32; f += 512) {
        int fl = f & 31;
        int nt = (f >> 5) & 3;
        int kk = f >> 7;
        unsigned lo = lds_off(&efT[(nt * 16 + (fl & 15)) * 528 + kk * 32 + ((fl >> 4) << 4)]);
        __bf16* dst = ef + ((size_t)(baseTile + nt) * 16 + kk) * 32 * 16 + (size_t)fl * 16;
        asm volatile("global_store_async_from_lds_b128 %0, %1, off"
                     :: "v"(dst), "v"(lo) : "memory");
        asm volatile("global_store_async_from_lds_b128 %0, %1, off offset:16"
                     :: "v"(dst), "v"(lo) : "memory");
    }
    asm volatile("s_wait_asynccnt 0x0" ::: "memory");
}

// ---------------------------------------------------------------------------
// Deterministic fixed-order reduction of per-WG sums -> pooled mean [4][512].
// ---------------------------------------------------------------------------
__global__ void reduce_pool(const float* __restrict__ partials, float* __restrict__ pooled) {
    int t = blockIdx.x * blockDim.x + threadIdx.x;   // 2048
    int b = t >> 9;
    int c = t & 511;
    float s = 0.0f;
    for (int i = 0; i < 256; ++i)
        s += partials[(size_t)(b * 256 + i) * 512 + c];
    pooled[t] = s * (1.0f / (Hh * Ww));
}

// ---------------------------------------------------------------------------
// Tiny SE MLP: gelu(pooled @ w1^T + b1) -> sigmoid(h @ w2^T + b2).
// ---------------------------------------------------------------------------
__global__ void attention_mlp(const float* __restrict__ pooled,
                              const float* __restrict__ w1, const float* __restrict__ b1,
                              const float* __restrict__ w2, const float* __restrict__ b2,
                              float* __restrict__ att) {
    __shared__ float ps[4 * 512];
    __shared__ float hs[4 * 64];
    int tid = threadIdx.x;   // 256
    for (int i = tid; i < 2048; i += 256) ps[i] = pooled[i];
    __syncthreads();
    {
        int b = tid >> 6, e = tid & 63;
        float s = b1[e];
        for (int c = 0; c < 512; ++c) s += ps[b * 512 + c] * w1[e * 512 + c];
        hs[tid] = gelu_exact(s);
    }
    __syncthreads();
    for (int i = tid; i < 2048; i += 256) {
        int b = i >> 9, c = i & 511;
        float s = b2[c];
        for (int e = 0; e < 64; ++e) s += hs[b * 64 + e] * w2[c * 64 + e];
        att[i] = 1.0f / (1.0f + expf(-s));
    }
}

// ---------------------------------------------------------------------------
// Pass 2: fusion GEMM (att folded into weights) + channel LayerNorm + GELU
// + residual.  B-fragments come pre-packed straight from global ef.
// ---------------------------------------------------------------------------
__global__ __launch_bounds__(512)
void fuse_ln_gemm(const __bf16* __restrict__ ef,
                  const __bf16* __restrict__ pwf,
                  const float* __restrict__ b_fuse,
                  const float* __restrict__ ln_g,
                  const float* __restrict__ ln_b,
                  const float* __restrict__ x,
                  float* __restrict__ out) {
    __shared__ float sumS[16 * 64];
    __shared__ float ssqS[16 * 64];
    __shared__ float muS[64], rsS[64];
    __shared__ float gS[512], bS[512], fS[512];

    const int tid  = threadIdx.x;
    const int wid  = blockIdx.x;
    const int b    = wid >> 8;
    const int h    = (wid >> 1) & 127;
    const int w0   = (wid & 1) * 64;
    const int wv   = tid >> 5;
    const int l    = tid & 31;
    const int lh   = l & 15;
    const int half = l >> 4;

    gS[tid] = ln_g[tid];
    bS[tid] = ln_b[tid];
    fS[tid] = b_fuse[tid];
    __syncthreads();

    const int baseTile = ((b * Hh + h) * Ww + w0) >> 4;
    v8f acc[2][4] = {};

    for (int kb = 0; kb < NKB; ++kb) {
        v16bf a0 = *(const v16bf*)(pwf + ((size_t)(((b * 32 + 2 * wv    ) * 16 + kb) * 32 + l)) * 16);
        v16bf a1 = *(const v16bf*)(pwf + ((size_t)(((b * 32 + 2 * wv + 1) * 16 + kb) * 32 + l)) * 16);
#pragma unroll
        for (int nt = 0; nt < 4; ++nt) {
            v16bf bf = *(const v16bf*)(ef + ((size_t)(baseTile + nt) * 16 + kb) * 32 * 16 + (size_t)l * 16);
            acc[0][nt] = __builtin_amdgcn_wmma_f32_16x16x32_bf16(false, a0, false, bf,
                             (short)0, acc[0][nt], false, false);
            acc[1][nt] = __builtin_amdgcn_wmma_f32_16x16x32_bf16(false, a1, false, bf,
                             (short)0, acc[1][nt], false, false);
        }
        if (kb + 1 < NKB)
            __builtin_prefetch(pwf + ((size_t)(((b * 32 + 2 * wv) * 16 + kb + 1) * 32 + l)) * 16, 0, 0);
    }

    // per-pixel sum / sumsq over this wave's 32 rows
#pragma unroll
    for (int nt = 0; nt < 4; ++nt) {
        float s1 = 0.0f, s2 = 0.0f;
#pragma unroll
        for (int tm = 0; tm < 2; ++tm) {
#pragma unroll
            for (int r = 0; r < 8; ++r) {
                int grow = (2 * wv + tm) * 16 + r + (half << 3);
                float z = acc[tm][nt][r] + fS[grow];
                s1 += z;
                s2 += z * z;
            }
        }
        s1 += __shfl_xor(s1, 16);
        s2 += __shfl_xor(s2, 16);
        if (half == 0) {
            sumS[wv * 64 + nt * 16 + lh] = s1;
            ssqS[wv * 64 + nt * 16 + lh] = s2;
        }
    }
    __syncthreads();
    if (tid < 64) {   // fixed-order cross-wave reduction -> deterministic
        float s1 = 0.0f, s2 = 0.0f;
        for (int w = 0; w < 16; ++w) { s1 += sumS[w * 64 + tid]; s2 += ssqS[w * 64 + tid]; }
        float mu  = s1 * (1.0f / 512.0f);
        float var = s2 * (1.0f / 512.0f) - mu * mu;
        muS[tid] = mu;
        rsS[tid] = rsqrtf(var + 1e-6f);
    }
    __syncthreads();

    // normalize + affine + GELU + residual, store fp32
#pragma unroll
    for (int tm = 0; tm < 2; ++tm) {
#pragma unroll
        for (int nt = 0; nt < 4; ++nt) {
            int col = nt * 16 + lh;
            float mu = muS[col], rs = rsS[col];
#pragma unroll
            for (int r = 0; r < 8; ++r) {
                int grow = (2 * wv + tm) * 16 + r + (half << 3);
                float z = acc[tm][nt][r] + fS[grow];
                float y = (z - mu) * rs * gS[grow] + bS[grow];
                size_t xi = (((size_t)b * Cch + grow) * Hh + h) * Ww + w0 + col;
                out[xi] = gelu_exact(y) + x[xi];
            }
        }
    }
}

// ---------------------------------------------------------------------------
extern "C" void kernel_launch(void* const* d_in, const int* in_sizes, int n_in,
                              void* d_out, int out_size, void* d_ws, size_t ws_size,
                              hipStream_t stream) {
    (void)in_sizes; (void)n_in; (void)out_size; (void)ws_size;
    const float* x        = (const float*)d_in[0];
    const float* w_branch = (const float*)d_in[1];
    const float* b_branch = (const float*)d_in[2];
    const float* bn_gamma = (const float*)d_in[3];
    const float* bn_beta  = (const float*)d_in[4];
    const float* w_att1   = (const float*)d_in[5];
    const float* b_att1   = (const float*)d_in[6];
    const float* w_att2   = (const float*)d_in[7];
    const float* b_att2   = (const float*)d_in[8];
    const float* w_fuse   = (const float*)d_in[9];
    const float* b_fuse   = (const float*)d_in[10];
    const float* ln_g     = (const float*)d_in[11];
    const float* ln_b     = (const float*)d_in[12];
    float* out = (float*)d_out;

    char* ws = (char*)d_ws;
    __bf16* ef       = (__bf16*)(ws);                                           // 64 MiB
    __bf16* pwb      = (__bf16*)(ws + (64u << 20));                             // 512 KiB
    __bf16* pwf      = (__bf16*)(ws + (64u << 20) + (512u << 10));              // 2 MiB
    float*  partials = (float*) (ws + (64u << 20) + (512u << 10) + (2u << 20)); // 2 MiB
    float*  pooled   = (float*) (ws + (64u << 20) + (512u << 10) + (4u << 20)); // 8 KiB
    float*  att      = (float*) (ws + (64u << 20) + (512u << 10) + (4u << 20) + 8192);

    pack_wbranch<<<64, 256, 0, stream>>>(w_branch, pwb);
    edge_branch_gemm<<<1024, 512, 0, stream>>>(x, pwb, b_branch, bn_gamma, bn_beta, ef, partials);
    reduce_pool<<<8, 256, 0, stream>>>(partials, pooled);
    attention_mlp<<<1, 256, 0, stream>>>(pooled, w_att1, b_att1, w_att2, b_att2, att);
    pack_wfuse<<<256, 256, 0, stream>>>(w_fuse, att, pwf);
    fuse_ln_gemm<<<1024, 512, 0, stream>>>(ef, pwf, b_fuse, ln_g, ln_b, x, out);
}
